// Net_2370821948180
// MI455X (gfx1250) — matrix operand order repaired and checked by reading, not compiled, not run on hardware
//
#include <hip/hip_runtime.h>
#include <hip/hip_bf16.h>

// ---------------------------------------------------------------------------
// 3-layer LSTM (T=2048, N=256, H=64) + MLP head, fused persistent kernel.
// grid = 8 workgroups x 512 threads (16 waves). Each WG owns 32 batch rows
// and runs the whole 3-layer recurrence on-chip: weights live in VGPRs as
// WMMA B-fragments, h/c live in LDS/registers, gates staged through LDS.
// Matrix math: v_wmma_f32_16x16x32_f16 (f16 in, f32 accumulate).
// Activations: hardware v_tanh_f32 when available (gfx1250 TRANS op),
// branch-free exp/rcp fallback otherwise.
// ---------------------------------------------------------------------------

typedef __attribute__((ext_vector_type(16))) _Float16 v16h;
typedef __attribute__((ext_vector_type(8)))  float    v8f;

#define TSTEPS 2048
#define NB     256
#define HID    64
#define ROWS   32          // batch rows per workgroup
#define NWG    (NB / ROWS) // 8 workgroups
#define NTHR   512         // 16 waves of 32
#define HSTR   68          // padded LDS row stride (halves) to spread banks

// ---- smem region map (bytes), total 53504 < 64KB ----
#define OFF_GATES 0                      // float [32][256]            32768
#define OFF_HPREV 32768                  // _Float16 [3][32][HSTR]     13056
#define OFF_WIH0  45824                  // float [256][4]              4096
#define OFF_BIAS  49920                  // float [3][256]              3072
#define OFF_XBUF  52992                  // float [32][4]                512
#define SMEM_SZ   53504
// head reuse: buf1 f32[32][256] @0 ; buf2 f16[32][256] @32768 ; h3 f32[32][128] @0

// ---- fast branch-free tanh: prefer CDNA5 hardware v_tanh_f32 ----
__device__ __forceinline__ float fast_tanh(float v) {
#if __has_builtin(__builtin_amdgcn_tanhf)
    return __builtin_amdgcn_tanhf(v);
#elif __has_builtin(__builtin_amdgcn_tanh_f32)
    return __builtin_amdgcn_tanh_f32(v);
#else
    // branch-free: tanh(|v|) = (1 - e^{-2|v|}) / (1 + e^{-2|v|}), restore sign
    const float a = __builtin_fabsf(v);
    const float e = __expf(-2.0f * a);
    const float r = (1.0f - e) * __frcp_rn(1.0f + e);
    return __builtin_copysignf(r, v);
#endif
}

__device__ __forceinline__ float sigm(float v) {
    // sigmoid(x) = 0.5 * tanh(x/2) + 0.5  (single TRANS op on CDNA5)
    return __builtin_fmaf(0.5f, fast_tanh(0.5f * v), 0.5f);
}

__device__ __forceinline__ v8f wmma_f16(v16h a, v16h b, v8f c) {
    // (neg_a, A, neg_b, B, c_mod, C, reuse_a, reuse_b)
    return __builtin_amdgcn_wmma_f32_16x16x32_f16(false, a, false, b, (short)0, c,
                                                  false, false);
}

// B fragment (32x16 f16): lane&15 = column n; lanes 0-15 hold K={0..7,16..23}, lanes 16-31 hold K={8..15,24..31}.
// B[k][n] = W[n_global][k]  (gates = h @ W^T), W is [256][64] row-major f32.
__device__ __forceinline__ v16h load_bfrag(const float* W, int ctg, int kf, int lane) {
    const int n     = ctg * 16 + (lane & 15);
    const int kbase = kf * 32 + ((lane >> 4) << 3);
    const float* p  = W + n * 64 + kbase;
    v16h b;
#pragma unroll
    for (int e = 0; e < 8; ++e) { b[e] = (_Float16)p[e]; b[8 + e] = (_Float16)p[16 + e]; }
    return b;
}

// A fragment (16x32 f16) from LDS h[32][HSTR]:
// lanes 0-15: K = base+{0..7,16..23}; lanes 16-31: K = base+{8..15,24..31}.
__device__ __forceinline__ v16h load_afrag(const _Float16* hsrc, int rt, int kf, int lane) {
    const int m   = rt * 16 + (lane & 15);
    const int off = (lane >> 4) * 8;
    const _Float16* p = hsrc + m * HSTR + kf * 32;
    v16h a;
#pragma unroll
    for (int e = 0; e < 8; ++e) {
        a[e]     = p[off + e];
        a[8 + e] = p[16 + off + e];
    }
    return a;
}

// C/D tile (16x16 f32): VGPR v -> row rt*16 + v + (lane>=16 ? 8 : 0), col = lane&15.
__device__ __forceinline__ void store_tile(float* gates, v8f acc, int rt, int ctg, int lane) {
    const int col   = ctg * 16 + (lane & 15);
    const int rbase = rt * 16 + ((lane >> 4) ? 8 : 0);
#pragma unroll
    for (int v = 0; v < 8; ++v) gates[(rbase + v) * 256 + col] = acc[v];
}

__global__ __launch_bounds__(NTHR) void lstm3_mlp_kernel(
    const float* __restrict__ x,    const float* __restrict__ h0,  const float* __restrict__ c0,
    const float* __restrict__ Wih0, const float* __restrict__ Whh0,
    const float* __restrict__ bih0, const float* __restrict__ bhh0,
    const float* __restrict__ Wih1, const float* __restrict__ Whh1,
    const float* __restrict__ bih1, const float* __restrict__ bhh1,
    const float* __restrict__ Wih2, const float* __restrict__ Whh2,
    const float* __restrict__ bih2, const float* __restrict__ bhh2,
    const float* __restrict__ W1, const float* __restrict__ b1,
    const float* __restrict__ W2, const float* __restrict__ b2,
    const float* __restrict__ W3, const float* __restrict__ b3,
    const float* __restrict__ W4, const float* __restrict__ b4,
    float* __restrict__ out)
{
    __shared__ __align__(16) unsigned char smem[SMEM_SZ];
    float*     gates = (float*)(smem + OFF_GATES);     // [32][256]
    _Float16*  hprev = (_Float16*)(smem + OFF_HPREV);  // [3][32][HSTR]
    float*     wih0s = (float*)(smem + OFF_WIH0);      // [256][4]
    float*     biass = (float*)(smem + OFF_BIAS);      // [3][256]
    float*     xbuf  = (float*)(smem + OFF_XBUF);      // [32][4]

    const int tid  = threadIdx.x;
    const int lane = tid & 31;     // wave32
    const int ctg  = tid >> 5;     // wave id == gate column tile (0..15)
    const int r0   = blockIdx.x * ROWS;

    // ---- persistent weight B-fragments in VGPRs: Whh0,Wih1,Whh1,Wih2,Whh2 ----
    const float* mats[5] = { Whh0, Wih1, Whh1, Wih2, Whh2 };
    v16h wB[5][2];
#pragma unroll
    for (int m = 0; m < 5; ++m)
#pragma unroll
        for (int kf = 0; kf < 2; ++kf) wB[m][kf] = load_bfrag(mats[m], ctg, kf, lane);

    // ---- LDS init: Wih0, fused biases ----
    for (int i = tid; i < 256 * 4; i += NTHR) wih0s[i] = Wih0[i];
    for (int i = tid; i < 3 * 256; i += NTHR) {
        const int l = i >> 8, n = i & 255;
        const float* bi = (l == 0) ? bih0 : (l == 1) ? bih1 : bih2;
        const float* bh = (l == 0) ? bhh0 : (l == 1) ? bhh1 : bhh2;
        biass[i] = bi[n] + bh[n];
    }

    // ---- state init: h -> LDS f16, c -> registers ----
    float c_reg[3][4];
#pragma unroll
    for (int l = 0; l < 3; ++l)
#pragma unroll
        for (int q = 0; q < 4; ++q) {
            const int p = q * NTHR + tid;      // 0..2047
            const int r = p >> 6, j = p & 63;
            hprev[l * (ROWS * HSTR) + r * HSTR + j] =
                (_Float16)h0[(l * NB + r0 + r) * HID + j];
            c_reg[l][q] = c0[(l * NB + r0 + r) * HID + j];
        }
    __syncthreads();

    const int whhIdx[3] = { 0, 2, 4 };
    const int wihIdx[3] = { 0, 1, 3 };   // [0] unused

    // =================== time recurrence ===================
    for (int t = 0; t < TSTEPS; ++t) {
        if (tid < ROWS * 4) xbuf[tid] = x[(t * NB + r0) * 4 + tid];
        if (t + 1 < TSTEPS && tid < 8)
            __builtin_prefetch(&x[((t + 1) * NB + r0) * 4 + tid * 16], 0, 0);

#pragma unroll
        for (int l = 0; l < 3; ++l) {
            const _Float16* hsrc = hprev + l * (ROWS * HSTR);

            v16h aH[2][2];
#pragma unroll
            for (int rt = 0; rt < 2; ++rt)
#pragma unroll
                for (int kf = 0; kf < 2; ++kf) aH[rt][kf] = load_afrag(hsrc, rt, kf, lane);

            v16h aX[2][2];
            if (l > 0) {
                const _Float16* xsrc = hprev + (l - 1) * (ROWS * HSTR);
#pragma unroll
                for (int rt = 0; rt < 2; ++rt)
#pragma unroll
                    for (int kf = 0; kf < 2; ++kf) aX[rt][kf] = load_afrag(xsrc, rt, kf, lane);
            }

#pragma unroll
            for (int rt = 0; rt < 2; ++rt) {
                v8f acc = {};
#pragma unroll
                for (int kf = 0; kf < 2; ++kf)
                    acc = wmma_f16(aH[rt][kf], wB[whhIdx[l]][kf], acc);
                if (l > 0) {
#pragma unroll
                    for (int kf = 0; kf < 2; ++kf)
                        acc = wmma_f16(aX[rt][kf], wB[wihIdx[l]][kf], acc);
                }
                store_tile(gates, acc, rt, ctg, lane);
            }
            __syncthreads();

            // fused gate nonlinearity + cell update; h back to LDS as f16
#pragma unroll
            for (int q = 0; q < 4; ++q) {
                const int p = q * NTHR + tid;
                const int r = p >> 6, j = p & 63;
                float gi = gates[r * 256 + j]       + biass[l * 256 + j];
                float gf = gates[r * 256 + 64 + j]  + biass[l * 256 + 64 + j];
                float gg = gates[r * 256 + 128 + j] + biass[l * 256 + 128 + j];
                float go = gates[r * 256 + 192 + j] + biass[l * 256 + 192 + j];
                if (l == 0) {
#pragma unroll
                    for (int k = 0; k < 4; ++k) {
                        const float xv = xbuf[r * 4 + k];
                        gi = fmaf(xv, wih0s[j * 4 + k],         gi);
                        gf = fmaf(xv, wih0s[(64  + j) * 4 + k], gf);
                        gg = fmaf(xv, wih0s[(128 + j) * 4 + k], gg);
                        go = fmaf(xv, wih0s[(192 + j) * 4 + k], go);
                    }
                }
                const float iS = sigm(gi), fS = sigm(gf);
                const float gT = fast_tanh(gg), oS = sigm(go);
                const float cn = fmaf(fS, c_reg[l][q], iS * gT);
                c_reg[l][q] = cn;
                hprev[l * (ROWS * HSTR) + r * HSTR + j] = (_Float16)(oS * fast_tanh(cn));
            }
            __syncthreads();
        }
    }

    // =================== MLP head on last h of layer 2 ===================
    const _Float16* hlast = hprev + 2 * (ROWS * HSTR);

    // h1 = relu(h @ W1^T + b1) -> gates region (f32 [32][256])
    for (int i = tid; i < ROWS * 256; i += NTHR) {
        const int r = i >> 8, n = i & 255;
        float s = b1[n];
        for (int k = 0; k < 64; ++k) s = fmaf((float)hlast[r * HSTR + k], W1[n * 64 + k], s);
        gates[r * 256 + n] = fmaxf(s, 0.0f);
    }
    __syncthreads();

    // h2 = relu(h1 @ W2^T + b2) -> f16 buf2 @32768 (hprev/wih0 regions dead)
    _Float16* buf2 = (_Float16*)(smem + 32768);
    for (int i = tid; i < ROWS * 256; i += NTHR) {
        const int r = i >> 8, n = i & 255;
        float s = b2[n];
        for (int k = 0; k < 256; ++k) s = fmaf(gates[r * 256 + k], W2[n * 256 + k], s);
        buf2[r * 256 + n] = (_Float16)fmaxf(s, 0.0f);
    }
    __syncthreads();

    // h3 = relu(h2 @ W3^T + b3) -> f32 @0 (gates region dead)
    float* h3 = (float*)smem;
    for (int i = tid; i < ROWS * 128; i += NTHR) {
        const int r = i >> 7, n = i & 127;
        float s = b3[n];
        for (int k = 0; k < 256; ++k) s = fmaf((float)buf2[r * 256 + k], W3[n * 256 + k], s);
        h3[r * 128 + n] = fmaxf(s, 0.0f);
    }
    __syncthreads();

    // out = h3 @ W4^T + b4   ([32][4] per WG)
    if (tid < ROWS * 4) {
        const int r = tid >> 2, n = tid & 3;
        float s = b4[n];
        for (int k = 0; k < 128; ++k) s = fmaf(h3[r * 128 + k], W4[n * 128 + k], s);
        out[(r0 + r) * 4 + n] = s;
    }
}

extern "C" void kernel_launch(void* const* d_in, const int* in_sizes, int n_in,
                              void* d_out, int out_size, void* d_ws, size_t ws_size,
                              hipStream_t stream) {
    (void)in_sizes; (void)n_in; (void)out_size; (void)d_ws; (void)ws_size;
    const float* x    = (const float*)d_in[0];
    const float* h0   = (const float*)d_in[1];
    const float* c0   = (const float*)d_in[2];
    const float* Wih0 = (const float*)d_in[3];
    const float* Whh0 = (const float*)d_in[4];
    const float* bih0 = (const float*)d_in[5];
    const float* bhh0 = (const float*)d_in[6];
    const float* Wih1 = (const float*)d_in[7];
    const float* Whh1 = (const float*)d_in[8];
    const float* bih1 = (const float*)d_in[9];
    const float* bhh1 = (const float*)d_in[10];
    const float* Wih2 = (const float*)d_in[11];
    const float* Whh2 = (const float*)d_in[12];
    const float* bih2 = (const float*)d_in[13];
    const float* bhh2 = (const float*)d_in[14];
    const float* W1   = (const float*)d_in[15];
    const float* b1   = (const float*)d_in[16];
    const float* W2   = (const float*)d_in[17];
    const float* b2   = (const float*)d_in[18];
    const float* W3   = (const float*)d_in[19];
    const float* b3   = (const float*)d_in[20];
    const float* W4   = (const float*)d_in[21];
    const float* b4   = (const float*)d_in[22];

    lstm3_mlp_kernel<<<dim3(NWG), dim3(NTHR), 0, stream>>>(
        x, h0, c0, Wih0, Whh0, bih0, bhh0, Wih1, Whh1, bih1, bhh1,
        Wih2, Whh2, bih2, bhh2, W1, b1, W2, b2, W3, b3, W4, b4,
        (float*)d_out);
}
